// CrossAttention_8933531976383
// MI455X (gfx1250) — matrix operand order repaired and checked by reading, not compiled, not run on hardware
//
#include <hip/hip_runtime.h>
#include <hip/hip_bf16.h>
#include <stdint.h>

#define BB 8
#define SS 2048
#define DD 1024
#define OO 1024

typedef __attribute__((ext_vector_type(16))) __bf16 v16bf;
typedef __attribute__((ext_vector_type(8)))  float  v8f;
typedef __attribute__((ext_vector_type(4)))  unsigned int u32x4;
typedef __attribute__((ext_vector_type(8)))  int i32x8;
typedef __attribute__((ext_vector_type(4)))  int i32x4;

union Frag16 {
    v16bf v;
    uint32_t u[8];
    uint4 q[2];
};

// ---------------------------------------------------------------------------
// Tensor Data Mover: issue a 2D bf16 tile load (rows x width) global -> LDS.
// D# bit layout per CDNA5 ISA §8.3/8.4 (2D tensor, groups 2/3 zero).
// 6-arg builtin form (this toolchain): (g0, g1, g2, g3, g4, cpol).
// ---------------------------------------------------------------------------
__device__ __forceinline__ void tdm_load_tile_2d(uint32_t lds_off,
                                                 const void* gaddr,
                                                 uint32_t width_elems,
                                                 uint32_t rows,
                                                 uint32_t stride_elems) {
    const uint64_t ga = (uint64_t)(uintptr_t)gaddr;
    u32x4 g0;
    g0[0] = 1u;                                   // count=1, user descriptor
    g0[1] = lds_off;                              // lds_addr (bytes)
    g0[2] = (uint32_t)(ga & 0xFFFFFFFFu);         // global_addr[31:0]
    g0[3] = (uint32_t)((ga >> 32) & 0x1FFFFFFu)   // global_addr[56:32]
            | (2u << 30);                         // type = 2 ("image")
    i32x8 g1;
    const uint32_t td0 = stride_elems;            // tensor_dim0 (>= tile width)
    const uint32_t td1 = rows;                    // tensor_dim1
    g1[0] = (int)(1u << 16);                      // data_size=1 (2 bytes)
    g1[1] = (int)((td0 & 0xFFFFu) << 16);         // tensor_dim0[15:0]
    g1[2] = (int)((td0 >> 16) | ((td1 & 0xFFFFu) << 16));
    g1[3] = (int)((td1 >> 16) | (width_elems << 16));  // tile_dim0
    g1[4] = (int)(rows & 0xFFFFu);                // tile_dim1 ; tile_dim2=0
    g1[5] = (int)stride_elems;                    // tensor_dim0_stride[31:0]
    g1[6] = 0;                                    // stride hi / dim1_stride lo
    g1[7] = 0;
    const i32x4 gz4 = {0, 0, 0, 0};
    const i32x8 gz8 = {0, 0, 0, 0, 0, 0, 0, 0};
    __builtin_amdgcn_tensor_load_to_lds(g0, g1, gz4, gz4, gz8, 0);
}

// ---------------------------------------------------------------------------
// fp32 -> bf16 cast
// ---------------------------------------------------------------------------
__global__ void cast_f32_bf16(const float* __restrict__ src,
                              __bf16* __restrict__ dst, int n) {
    int i = blockIdx.x * blockDim.x + threadIdx.x;
    int stride = gridDim.x * blockDim.x;
    for (; i < n; i += stride) dst[i] = (__bf16)src[i];
}

// ---------------------------------------------------------------------------
// fp32 [R,C] -> bf16 transposed [C,R]  (one-time weight transpose, 1M elems)
// ---------------------------------------------------------------------------
__global__ void transpose_cast_f32_bf16(const float* __restrict__ src,
                                        __bf16* __restrict__ dst, int R,
                                        int C) {
    int i = blockIdx.x * blockDim.x + threadIdx.x;
    const int n = R * C;
    const int stride = gridDim.x * blockDim.x;
    for (; i < n; i += stride) {
        const int r = i / C;
        const int c = i - r * C;
        dst[(long)c * R + r] = (__bf16)src[i];
    }
}

// ---------------------------------------------------------------------------
// bf16 GEMM with transposed B operand:
//   C[b] (MxN) = A[b] (MxK, row major, lda=K) * BT[b]^T  (BT is [N,K] row
//   major, ldb given) -- contraction axis contiguous for BOTH operands.
// Block tile 128(M) x 64(N); 8 waves, each computing 32x32 (2x2 WMMA tiles).
// The 64x32 BT tile is DMA'd k-contiguous into LDS by the Tensor Data Mover
// (double buffered, TENSORcnt); B fragments are two ds_load_b128 per lane.
// TRANS_OUT: store C transposed (C[n][m], bf16) -- used to emit v_proj^T.
// OUT_F32: f32 store with row stride ldc and column offset c_off (d_out).
// ---------------------------------------------------------------------------
template <bool OUT_F32, bool TRANS_OUT>
__global__ __launch_bounds__(256) void gemm_bf16(
    const __bf16* __restrict__ A, const __bf16* __restrict__ BT,
    void* __restrict__ Cout, int K, int ldb,
    long strideA, long strideB, long strideC, int ldc, int c_off) {
    __shared__ __bf16 Bt[2][64 * 32];  // [n][k] k-contiguous, double buffer

    const int b = blockIdx.z;
    const __bf16* Ab = A + (long)b * strideA;
    const __bf16* Bb = BT + (long)b * strideB;

    const int tid = threadIdx.x;
    const int lane = tid & 31;
    const int wave = tid >> 5;
    const int half = lane >> 4;   // 0 | 1
    const int l16  = lane & 15;
    const int wm = wave & 3;      // 4 waves along M
    const int wn = wave >> 2;     // 2 waves along N

    const int blockM = blockIdx.y * 128;
    const int blockN = blockIdx.x * 64;

    const uint32_t lds_buf0 =
        (uint32_t)(uintptr_t)&Bt[0][0];  // low 32 bits of flat == LDS offset
    const uint32_t lds_buf1 = (uint32_t)(uintptr_t)&Bt[1][0];

    v8f acc[2][2] = {};

    const int ntiles = K >> 5;  // K / 32

    // Prologue: DMA tile 0 into buffer 0 (one wave drives the TDM).
    if (wave == 0) {
        tdm_load_tile_2d(lds_buf0, Bb + (long)blockN * ldb, 32u, 64u,
                         (uint32_t)ldb);
    }

    for (int kt = 0; kt < ntiles; ++kt) {
        const int cur = kt & 1;
        const int k0 = kt << 5;
        if (wave == 0) {
            if (kt + 1 < ntiles) {
                tdm_load_tile_2d(cur ? lds_buf0 : lds_buf1,
                                 Bb + (long)blockN * ldb + k0 + 32, 32u, 64u,
                                 (uint32_t)ldb);
                __builtin_amdgcn_s_wait_tensorcnt(1);  // tile kt complete
            } else {
                __builtin_amdgcn_s_wait_tensorcnt(0);
            }
        }
        __syncthreads();  // LDS tile kt visible to all waves

        // A fragments (16x32 bf16, ISA K-striping): per lane two 16B loads
        Frag16 af[2];
#pragma unroll
        for (int mt = 0; mt < 2; ++mt) {
            const int m = blockM + wm * 32 + mt * 16 + l16;
            const __bf16* arow = Ab + (long)m * K + k0;
            __builtin_prefetch(arow + 32, 0, 0);  // global_prefetch_b8
            af[mt].q[0] = *(const uint4*)(arow + half * 8);
            af[mt].q[1] = *(const uint4*)(arow + 16 + half * 8);
        }
        // B fragments from LDS: lane covers k = half*8 .. half*8+7 and
        // 16+half*8 .. 16+half*8+7 -> two ds_load_b128.
        Frag16 bfr[2];
        const __bf16* L = &Bt[cur][0];
#pragma unroll
        for (int nt = 0; nt < 2; ++nt) {
            const int nl = wn * 32 + nt * 16 + l16;
            const __bf16* brow = L + nl * 32 + half * 8;
            bfr[nt].q[0] = *(const uint4*)(brow);
            bfr[nt].q[1] = *(const uint4*)(brow + 16);
        }
#pragma unroll
        for (int mt = 0; mt < 2; ++mt)
#pragma unroll
            for (int nt = 0; nt < 2; ++nt)
                acc[mt][nt] = __builtin_amdgcn_wmma_f32_16x16x32_bf16(
                    false, af[mt].v, false, bfr[nt].v, (short)0, acc[mt][nt],
                    false, false);

        __syncthreads();  // all reads of buffer `cur` done before TDM reuse
    }

    // epilogue: C/D layout -> VGPR r: lanes 0-15 M=r, lanes 16-31 M=r+8
#pragma unroll
    for (int mt = 0; mt < 2; ++mt) {
#pragma unroll
        for (int nt = 0; nt < 2; ++nt) {
            const int n = blockN + wn * 32 + nt * 16 + l16;
#pragma unroll
            for (int r = 0; r < 8; ++r) {
                const int m = blockM + wm * 32 + mt * 16 + r + half * 8;
                const float val = acc[mt][nt][r];
                if (TRANS_OUT) {
                    __bf16* C = (__bf16*)Cout + (long)b * strideC;
                    C[(long)n * ldc + m] = (__bf16)val;
                } else if (OUT_F32) {
                    float* C = (float*)Cout + (long)b * strideC;
                    C[(long)m * ldc + c_off + n] = val;
                } else {
                    __bf16* C = (__bf16*)Cout + (long)b * strideC;
                    C[(long)m * ldc + n] = (__bf16)val;
                }
            }
        }
    }
}

// ---------------------------------------------------------------------------
// attn[b, q, k] = sum_o q_proj[b,q,o] * k_proj[b,k,o]
// Both operands contraction-contiguous -> direct global 16B fragment loads,
// fully pipelined global_load_b128 + wmma. f32 logits out.
// ---------------------------------------------------------------------------
__global__ __launch_bounds__(256) void attn_logits(
    const __bf16* __restrict__ qp, const __bf16* __restrict__ kp,
    float* __restrict__ attn) {
    const int b = blockIdx.z;
    const __bf16* qb = qp + (long)b * SS * OO;
    const __bf16* kb = kp + (long)b * SS * OO;
    float* ab = attn + (long)b * SS * SS;

    const int tid = threadIdx.x;
    const int lane = tid & 31;
    const int wave = tid >> 5;
    const int half = lane >> 4;
    const int l16  = lane & 15;
    const int wm = wave & 3;
    const int wn = wave >> 2;

    const int blockM = blockIdx.y * 128;
    const int blockN = blockIdx.x * 64;

    v8f acc[2][2] = {};

    for (int k0 = 0; k0 < OO; k0 += 32) {
        Frag16 af[2], bfr[2];
#pragma unroll
        for (int mt = 0; mt < 2; ++mt) {
            const int m = blockM + wm * 32 + mt * 16 + l16;
            const __bf16* arow = qb + (long)m * OO + k0;
            af[mt].q[0] = *(const uint4*)(arow + half * 8);
            af[mt].q[1] = *(const uint4*)(arow + 16 + half * 8);
        }
#pragma unroll
        for (int nt = 0; nt < 2; ++nt) {
            const int n = blockN + wn * 32 + nt * 16 + l16;
            const __bf16* brow = kb + (long)n * OO + k0 + half * 8;
            bfr[nt].q[0] = *(const uint4*)(brow);
            bfr[nt].q[1] = *(const uint4*)(brow + 16);
        }
#pragma unroll
        for (int mt = 0; mt < 2; ++mt)
#pragma unroll
            for (int nt = 0; nt < 2; ++nt)
                acc[mt][nt] = __builtin_amdgcn_wmma_f32_16x16x32_bf16(
                    false, af[mt].v, false, bfr[nt].v, (short)0, acc[mt][nt],
                    false, false);
    }

#pragma unroll
    for (int mt = 0; mt < 2; ++mt)
#pragma unroll
        for (int nt = 0; nt < 2; ++nt) {
            const int n = blockN + wn * 32 + nt * 16 + l16;
#pragma unroll
            for (int r = 0; r < 8; ++r) {
                const int m = blockM + wm * 32 + mt * 16 + r + half * 8;
                ab[(long)m * SS + n] = acc[mt][nt][r];
            }
        }
}

// ---------------------------------------------------------------------------
// Row softmax over key axis; writes TRANSPOSED bf16 attn_w^T[b, k, q] so the
// context GEMM reads its contraction axis (q) contiguously.
// One 256-thread block per (b, q) row of 2048 keys.
// ---------------------------------------------------------------------------
__global__ __launch_bounds__(256) void softmax_T(
    const float* __restrict__ attn, __bf16* __restrict__ attnT) {
    __shared__ float red[256];
    const int row = blockIdx.x;  // b*SS + iq
    const int b = row / SS;
    const int iq = row - b * SS;
    const float* arow = attn + (long)row * SS;

    float vals[8];
    float mx = -3.4e38f;
#pragma unroll
    for (int j = 0; j < 8; ++j) {
        vals[j] = arow[threadIdx.x + j * 256];
        mx = fmaxf(mx, vals[j]);
    }
    red[threadIdx.x] = mx;
    __syncthreads();
    for (int s = 128; s > 0; s >>= 1) {
        if (threadIdx.x < s)
            red[threadIdx.x] = fmaxf(red[threadIdx.x], red[threadIdx.x + s]);
        __syncthreads();
    }
    mx = red[0];
    __syncthreads();

    float sum = 0.f;
#pragma unroll
    for (int j = 0; j < 8; ++j) {
        vals[j] = __expf(vals[j] - mx);
        sum += vals[j];
    }
    red[threadIdx.x] = sum;
    __syncthreads();
    for (int s = 128; s > 0; s >>= 1) {
        if (threadIdx.x < s) red[threadIdx.x] += red[threadIdx.x + s];
        __syncthreads();
    }
    const float inv = 1.f / red[0];

#pragma unroll
    for (int j = 0; j < 8; ++j) {
        const int ik = threadIdx.x + j * 256;
        attnT[((long)b * SS + ik) * SS + iq] = (__bf16)(vals[j] * inv);
    }
}

// ---------------------------------------------------------------------------
// out[b, s, 0:D] = q[b, s, :]
// ---------------------------------------------------------------------------
__global__ void copy_q(const float* __restrict__ q, float* __restrict__ out) {
    const long total = (long)BB * SS * DD;
    long i = (long)blockIdx.x * blockDim.x + threadIdx.x;
    const long stride = (long)gridDim.x * blockDim.x;
    for (; i < total; i += stride) {
        const long d = i % DD;
        const long s = i / DD;  // flattened (b*SS + s)
        out[s * (DD + OO) + d] = q[i];
    }
}

// ---------------------------------------------------------------------------
extern "C" void kernel_launch(void* const* d_in, const int* in_sizes, int n_in,
                              void* d_out, int out_size, void* d_ws,
                              size_t ws_size, hipStream_t stream) {
    const float* q  = (const float*)d_in[0];
    const float* k  = (const float*)d_in[1];
    const float* Wq = (const float*)d_in[2];
    const float* Wk = (const float*)d_in[3];
    const float* Wv = (const float*)d_in[4];
    float* out = (float*)d_out;
    char* ws = (char*)d_ws;
    const size_t MB = (size_t)1 << 20;

    __bf16* qh    = (__bf16*)(ws + 0 * MB);     // 32 MB
    __bf16* kh    = (__bf16*)(ws + 32 * MB);    // 32 MB
    __bf16* WqT   = (__bf16*)(ws + 64 * MB);    // 2 MB  (W^T [O,D])
    __bf16* WkT   = (__bf16*)(ws + 66 * MB);    // 2 MB
    __bf16* WvT   = (__bf16*)(ws + 68 * MB);    // 2 MB
    __bf16* qp    = (__bf16*)(ws + 72 * MB);    // 32 MB ([B*S, O])
    __bf16* kp    = (__bf16*)(ws + 104 * MB);   // 32 MB
    __bf16* vpT   = (__bf16*)(ws + 136 * MB);   // 32 MB (v_proj^T [O, B*S])
    float*  attn  = (float*)(ws + 168 * MB);    // 128 MB
    __bf16* attnT = (__bf16*)(ws + 296 * MB);   // 64 MB

    const int nqk = BB * SS * DD;  // 16M
    cast_f32_bf16<<<4096, 256, 0, stream>>>(q, qh, nqk);
    cast_f32_bf16<<<4096, 256, 0, stream>>>(k, kh, nqk);
    transpose_cast_f32_bf16<<<1024, 256, 0, stream>>>(Wq, WqT, DD, OO);
    transpose_cast_f32_bf16<<<1024, 256, 0, stream>>>(Wk, WkT, DD, OO);
    transpose_cast_f32_bf16<<<1024, 256, 0, stream>>>(Wv, WvT, DD, OO);

    // Projections: [16384 x 1024] * W^T^T, bf16 out (v written transposed)
    dim3 gp(OO / 64, (BB * SS) / 128, 1);
    gemm_bf16<false, false><<<gp, 256, 0, stream>>>(qh, WqT, qp, DD, DD, 0, 0,
                                                    0, OO, 0);
    gemm_bf16<false, false><<<gp, 256, 0, stream>>>(kh, WkT, kp, DD, DD, 0, 0,
                                                    0, OO, 0);
    gemm_bf16<false, true><<<gp, 256, 0, stream>>>(kh, WvT, vpT, DD, DD, 0, 0,
                                                   0, BB * SS, 0);

    // Logits q_proj @ k_proj^T -> f32 [B, S, S]
    dim3 ga(SS / 64, SS / 128, BB);
    attn_logits<<<ga, 256, 0, stream>>>(qp, kp, attn);

    // Softmax over keys, emit transposed bf16 weights
    softmax_T<<<BB * SS, 256, 0, stream>>>(attn, attnT);

    // Context: per batch attn_w^T[S,S] @ v_proj[S,O] -> f32 into out[..., D:]
    // A = attnT (lda = K = S), BT = vpT (ldb = B*S, batch offset = b*S cols)
    dim3 gc(OO / 64, SS / 128, BB);
    gemm_bf16<true, false><<<gc, 256, 0, stream>>>(
        attnT, vpT, out, SS, BB * SS, (long)SS * SS, (long)SS,
        (long)SS * (DD + OO), DD + OO, DD);

    // out[..., :D] = q
    copy_q<<<4096, 256, 0, stream>>>(q, out);
}
